// SelfTransformer_72739566125136
// MI455X (gfx1250) — matrix operand order repaired and checked by reading, not compiled, not run on hardware
//
#include <hip/hip_runtime.h>

// Problem constants
#define BB   4
#define HH   64
#define WW   64
#define DD   384
#define MM   12
#define PP   9
#define DHH  32
#define DFFF 1152
#define NN   4096   // H*W
#define BN   (BB * NN)

typedef __bf16 bf16;
typedef __attribute__((ext_vector_type(16))) __bf16 v16bf;
typedef __attribute__((ext_vector_type(8)))  float  v8f;

#if __has_builtin(__builtin_amdgcn_global_load_async_to_lds_b128)
#define HAVE_ASYNC_LDS 1
typedef int v4i_vec __attribute__((vector_size(16)));
typedef __attribute__((address_space(1))) v4i_vec as1_v4i;   // global
typedef __attribute__((address_space(3))) v4i_vec as3_v4i;   // LDS
#else
#define HAVE_ASYNC_LDS 0
#endif

__device__ __forceinline__ void wait_async0() {
#if __has_builtin(__builtin_amdgcn_s_wait_asynccnt)
  __builtin_amdgcn_s_wait_asynccnt(0);
#else
  asm volatile("s_wait_asynccnt 0x0" ::: "memory");
#endif
}

// ---------------------------------------------------------------------------
// f32 -> bf16 weight conversion
// ---------------------------------------------------------------------------
__global__ __launch_bounds__(256) void f2bf_kernel(const float* __restrict__ s,
                                                   bf16* __restrict__ d, int n) {
  int i = blockIdx.x * 256 + threadIdx.x;
  if (i < n) d[i] = (bf16)s[i];
}

// ---------------------------------------------------------------------------
// Sine positional encoding + transpose x[B,D,N] -> src[B,N,D]
// ---------------------------------------------------------------------------
__global__ __launch_bounds__(384)
void pos_embed_kernel(const float* __restrict__ x, const float* __restrict__ qm,
                      const float* __restrict__ le, float* __restrict__ srcF,
                      bf16* __restrict__ srcB, bf16* __restrict__ qposB) {
  const int pix = blockIdx.x;          // b*N + n
  const int b   = pix / NN;
  const int n   = pix % NN;
  const int h   = n / WW;
  const int wq  = n % WW;
  const int d   = threadIdx.x;

  __shared__ float ynorm, xnorm;
  if (d == 0) {
    const float twopi = 6.28318530717958647692f;
    const float* mb = qm + (size_t)b * NN;
    float cum = 0.f, tot = 0.f;
    for (int hh = 0; hh < HH; ++hh) {
      float nm = (mb[hh * WW + wq] == 255.0f) ? 0.f : 1.f;
      tot += nm;
      if (hh <= h) cum += nm;
    }
    ynorm = cum / (tot + 1e-6f) * twopi;
    const float* mr = mb + h * WW;
    float cumx = 0.f, totx = 0.f;
    for (int ww = 0; ww < WW; ++ww) {
      float nm = (mr[ww] == 255.0f) ? 0.f : 1.f;
      totx += nm;
      if (ww <= wq) cumx += nm;
    }
    xnorm = cumx / (totx + 1e-6f) * twopi;
  }
  __syncthreads();

  const int nf = DD / 2;               // 192
  float c = (d < nf) ? ynorm : xnorm;
  int   j = d % nf;
  float t = __powf(10000.0f, (float)(2 * (j >> 1)) / (float)nf);
  float pos = (j & 1) ? __cosf(c / t) : __sinf(c / t);

  float xv = x[((size_t)b * DD + d) * NN + n];
  size_t o = ((size_t)(b * NN + n)) * DD + d;
  srcF[o]  = xv;
  srcB[o]  = (bf16)xv;
  qposB[o] = (bf16)(xv + pos + le[d]);
}

// ---------------------------------------------------------------------------
// bf16 WMMA GEMM:  C[Mrows,Nout] = A[Mrows,K] * W[Nout,K]^T (+bias)
// Block: 256 threads = 8 waves in a 2(M) x 4(N) grid; block tile 64M x 128N;
// each wave computes a 32x32 register tile => 4 v_wmma per K-step (K step 32).
// Tiles staged in LDS via async B128 copies when available.
// epilogue: 0 = f32 out (+bias)   1 = f32 out, rows with qmask==255 -> 0
//           2 = relu -> bf16 out
// ---------------------------------------------------------------------------
__global__ __launch_bounds__(256)
void gemm_bf16_wmma(const bf16* __restrict__ A, const bf16* __restrict__ Wt,
                    const float* __restrict__ bias, int Mrows, int Nout, int K,
                    float* __restrict__ outF, bf16* __restrict__ outB,
                    int epilogue, const float* __restrict__ qmask) {
  __shared__ uint4 lds_a[64][4];    // 64 rows  x 32 bf16
  __shared__ uint4 lds_b[128][4];   // 128 rows x 32 bf16

  const int ntiles = (Nout + 127) >> 7;
  const int bm = blockIdx.x / ntiles;
  const int bn = blockIdx.x % ntiles;
  const int mbase = bm << 6;
  const int nbase = bn << 7;
  const int tid  = threadIdx.x;
  const int wv   = tid >> 5;
  const int lane = tid & 31;
  const int wm   = wv >> 2;             // 0..1
  const int wn   = wv & 3;              // 0..3
  const int half = lane >> 4;           // 0: lanes 0-15, 1: lanes 16-31
  const int l16  = lane & 15;

  // staging coordinates: A = 256 uint4 (1/thread), B = 512 uint4 (2/thread)
  const int ar = tid >> 2;              // 0..63
  const int aq = tid & 3;               // uint4 within 32-elem row chunk
  const bool fullN = (nbase + 128 <= Nout);

  v8f acc00 = {}, acc01 = {}, acc10 = {}, acc11 = {};

  for (int kb = 0; kb < K; kb += 32) {
#if HAVE_ASYNC_LDS
    if (fullN) {
      __builtin_amdgcn_global_load_async_to_lds_b128(
          (as1_v4i*)(A + (size_t)(mbase + ar) * K + kb + aq * 8),
          (as3_v4i*)&lds_a[ar][aq], 0, 0);
      __builtin_amdgcn_global_load_async_to_lds_b128(
          (as1_v4i*)(Wt + (size_t)(nbase + ar) * K + kb + aq * 8),
          (as3_v4i*)&lds_b[ar][aq], 0, 0);
      __builtin_amdgcn_global_load_async_to_lds_b128(
          (as1_v4i*)(Wt + (size_t)(nbase + 64 + ar) * K + kb + aq * 8),
          (as3_v4i*)&lds_b[64 + ar][aq], 0, 0);
      wait_async0();
    } else
#endif
    {
      uint4 av = *(const uint4*)(A + (size_t)(mbase + ar) * K + kb + aq * 8);
      uint4 bv0, bv1;
      {
        int r0 = nbase + ar, r1 = nbase + 64 + ar;
        int c0 = (r0 < Nout) ? r0 : 0;
        int c1 = (r1 < Nout) ? r1 : 0;
        bv0 = *(const uint4*)(Wt + (size_t)c0 * K + kb + aq * 8);
        bv1 = *(const uint4*)(Wt + (size_t)c1 * K + kb + aq * 8);
        if (r0 >= Nout) { bv0.x = 0; bv0.y = 0; bv0.z = 0; bv0.w = 0; }
        if (r1 >= Nout) { bv1.x = 0; bv1.y = 0; bv1.z = 0; bv1.w = 0; }
      }
      lds_a[ar][aq]      = av;
      lds_b[ar][aq]      = bv0;
      lds_b[64 + ar][aq] = bv1;
    }
    __syncthreads();

    // Fragments per ISA layouts.
    // A 16x32: lane row = l16; lanes<16 hold K{0..7,16..23}, else K{8..15,24..31}
    const uint4* ar0 = lds_a[(wm << 5) + l16];
    const uint4* ar1 = lds_a[(wm << 5) + 16 + l16];
    // B 32x16 (=W^T): lane col = l16; lanes<16 hold K{0..7,16..23}, else K{8..15,24..31}
    const uint4* br0 = lds_b[(wn << 5) + l16];
    const uint4* br1 = lds_b[(wn << 5) + 16 + l16];

    union { uint4 u[2]; v16bf v; } a0, a1, b0, b1;
    a0.u[0] = ar0[half];     a0.u[1] = ar0[2 + half];
    a1.u[0] = ar1[half];     a1.u[1] = ar1[2 + half];
    b0.u[0] = br0[half];     b0.u[1] = br0[2 + half];
    b1.u[0] = br1[half];     b1.u[1] = br1[2 + half];

    acc00 = __builtin_amdgcn_wmma_f32_16x16x32_bf16(false, a0.v, false, b0.v,
                                                    (short)0, acc00, false, false);
    acc01 = __builtin_amdgcn_wmma_f32_16x16x32_bf16(false, a0.v, false, b1.v,
                                                    (short)0, acc01, false, false);
    acc10 = __builtin_amdgcn_wmma_f32_16x16x32_bf16(false, a1.v, false, b0.v,
                                                    (short)0, acc10, false, false);
    acc11 = __builtin_amdgcn_wmma_f32_16x16x32_bf16(false, a1.v, false, b1.v,
                                                    (short)0, acc11, false, false);
    __syncthreads();
  }

  // ---- epilogue ------------------------------------------------------------
  const int colbase = nbase + (wn << 5) + l16;
  const int rowbase = mbase + (wm << 5) + (half << 3);
  v8f* accs[2][2] = {{&acc00, &acc01}, {&acc10, &acc11}};
#pragma unroll
  for (int i = 0; i < 2; ++i) {
#pragma unroll
    for (int j = 0; j < 2; ++j) {
      int col = colbase + j * 16;
      if (col >= Nout) continue;
      v8f& acc = *accs[i][j];
      float bvs = bias ? bias[col] : 0.0f;
#pragma unroll
      for (int v = 0; v < 8; ++v) {
        int row = rowbase + i * 16 + v;
        float val = acc[v] + bvs;
        if (epilogue == 1 && qmask[row] == 255.0f) val = 0.0f;
        if (epilogue == 2) {
          val = fmaxf(val, 0.0f);
          outB[(size_t)row * Nout + col] = (bf16)val;
        } else {
          outF[(size_t)row * Nout + col] = val;
        }
      }
    }
  }
}

// ---------------------------------------------------------------------------
// Softmax over P=9 per (b,n,m), in place
// ---------------------------------------------------------------------------
__global__ __launch_bounds__(256) void softmax_p_kernel(float* __restrict__ aw) {
  int i = blockIdx.x * 256 + threadIdx.x;
  if (i >= BN * MM) return;
  float* a = aw + (size_t)i * PP;
  float mx = a[0];
#pragma unroll
  for (int p = 1; p < PP; ++p) mx = fmaxf(mx, a[p]);
  float e[PP], s = 0.f;
#pragma unroll
  for (int p = 0; p < PP; ++p) { e[p] = __expf(a[p] - mx); s += e[p]; }
  float inv = 1.0f / s;
#pragma unroll
  for (int p = 0; p < PP; ++p) a[p] = e[p] * inv;
}

// ---------------------------------------------------------------------------
// Deformable sampling: one wave per (b,n,m); lane = head channel (DH=32)
// ---------------------------------------------------------------------------
__global__ __launch_bounds__(256)
void msda_gather_kernel(const float* __restrict__ valF, const float* __restrict__ soF,
                        const float* __restrict__ awF, bf16* __restrict__ sampB) {
  const int wid  = blockIdx.x * 8 + (threadIdx.x >> 5);
  const int lane = threadIdx.x & 31;
  const int m  = wid % MM;
  const int bn = wid / MM;
  const int n  = bn % NN;
  const int b  = bn / NN;
  const int qw = n % WW, qh = n / WW;
  const float refx = (qw + 0.5f) / (float)WW;
  const float refy = (qh + 0.5f) / (float)HH;

  const float* so = soF + (((size_t)bn) * MM + m) * (PP * 2);
  const float* aw = awF + (((size_t)bn) * MM + m) * PP;
  const float* vb = valF + (size_t)b * NN * DD + m * DHH + lane;

  float acc = 0.f;
#pragma unroll
  for (int p = 0; p < PP; ++p) {
    float lx = refx + so[2 * p]     / (float)WW;
    float ly = refy + so[2 * p + 1] / (float)HH;
    float px = lx * WW - 0.5f;
    float py = ly * HH - 0.5f;
    float x0 = floorf(px), y0 = floorf(py);
    float fx = px - x0, fy = py - y0;
    int xi = (int)x0, yi = (int)y0;
    float a = aw[p];
    float wts[4] = {(1.f - fx) * (1.f - fy), fx * (1.f - fy),
                    (1.f - fx) * fy,         fx * fy};
    const int dxs[4] = {0, 1, 0, 1}, dys[4] = {0, 0, 1, 1};
#pragma unroll
    for (int c = 0; c < 4; ++c) {
      int xx = xi + dxs[c], yy = yi + dys[c];
      if (xx >= 0 && xx < WW && yy >= 0 && yy < HH)
        acc += a * wts[c] * vb[(size_t)(yy * WW + xx) * DD];
    }
  }
  sampB[((size_t)bn) * DD + m * DHH + lane] = (bf16)acc;
}

// ---------------------------------------------------------------------------
// LayerNorm(in1 + in2) * g + b -> outF (optionally transposed) + optional bf16
// ---------------------------------------------------------------------------
__global__ __launch_bounds__(384)
void layernorm_kernel(const float* __restrict__ in1, const float* __restrict__ in2,
                      const float* __restrict__ g, const float* __restrict__ bta,
                      float* __restrict__ outF, bf16* __restrict__ outB,
                      int transpose_out) {
  const int rown = blockIdx.x;   // b*N + n
  const int d = threadIdx.x;
  __shared__ float s[512];
  const size_t base = (size_t)rown * DD;
  float v = in1[base + d] + in2[base + d];
  s[d] = v;
  if (d < 128) s[d + 384] = 0.f;
  __syncthreads();
  for (int off = 256; off >= 1; off >>= 1) {
    if (d < off) s[d] += s[d + off];
    __syncthreads();
  }
  float mu = s[0] / (float)DD;
  __syncthreads();
  float dv = v - mu;
  s[d] = dv * dv;
  if (d < 128) s[d + 384] = 0.f;
  __syncthreads();
  for (int off = 256; off >= 1; off >>= 1) {
    if (d < off) s[d] += s[d + off];
    __syncthreads();
  }
  float var = s[0] / (float)DD;
  float r = rsqrtf(var + 1e-5f);
  float out = dv * r * g[d] + bta[d];
  if (transpose_out) {
    int b = rown / NN, n = rown % NN;
    outF[((size_t)b * DD + d) * NN + n] = out;
  } else {
    outF[base + d] = out;
  }
  if (outB) outB[base + d] = (bf16)out;
}

// ---------------------------------------------------------------------------
extern "C" void kernel_launch(void* const* d_in, const int* in_sizes, int n_in,
                              void* d_out, int out_size, void* d_ws, size_t ws_size,
                              hipStream_t stream) {
  (void)in_sizes; (void)n_in; (void)out_size; (void)ws_size;
  const float* x     = (const float*)d_in[0];
  const float* qm    = (const float*)d_in[1];
  const float* so_w  = (const float*)d_in[2];
  const float* so_b  = (const float*)d_in[3];
  const float* aw_w  = (const float*)d_in[4];
  const float* aw_b  = (const float*)d_in[5];
  const float* vp_w  = (const float*)d_in[6];
  const float* vp_b  = (const float*)d_in[7];
  const float* op_w  = (const float*)d_in[8];
  const float* op_b  = (const float*)d_in[9];
  const float* ln1_g = (const float*)d_in[10];
  const float* ln1_b = (const float*)d_in[11];
  const float* w1    = (const float*)d_in[12];
  const float* w2    = (const float*)d_in[13];
  const float* ln2_g = (const float*)d_in[14];
  const float* ln2_b = (const float*)d_in[15];
  const float* le    = (const float*)d_in[16];
  float* out = (float*)d_out;

  char* ws = (char*)d_ws;
  size_t off = 0;
  auto alloc = [&](size_t bytes) -> void* {
    void* p = ws + off;
    off = (off + bytes + 255) & ~(size_t)255;
    return p;
  };

  const int SO_N = MM * PP * 2;   // 216
  const int AW_N = MM * PP;       // 108

  bf16* sowB  = (bf16*)alloc((size_t)SO_N * DD * 2);
  bf16* awwB  = (bf16*)alloc((size_t)AW_N * DD * 2);
  bf16* vpwB  = (bf16*)alloc((size_t)DD * DD * 2);
  bf16* opwB  = (bf16*)alloc((size_t)DD * DD * 2);
  bf16* w1B   = (bf16*)alloc((size_t)DFFF * DD * 2);
  bf16* w2B   = (bf16*)alloc((size_t)DD * DFFF * 2);
  float* srcF = (float*)alloc((size_t)BN * DD * 4);
  bf16* srcB  = (bf16*)alloc((size_t)BN * DD * 2);
  bf16* qposB = (bf16*)alloc((size_t)BN * DD * 2);
  float* valF = (float*)alloc((size_t)BN * DD * 4);
  float* soF  = (float*)alloc((size_t)BN * SO_N * 4);
  float* awF  = (float*)alloc((size_t)BN * AW_N * 4);
  bf16* sampB = (bf16*)alloc((size_t)BN * DD * 2);
  float* attnF= (float*)alloc((size_t)BN * DD * 4);
  float* q1F  = (float*)alloc((size_t)BN * DD * 4);
  bf16* q1B   = (bf16*)alloc((size_t)BN * DD * 2);
  bf16* hB    = (bf16*)alloc((size_t)BN * DFFF * 2);
  float* ffnF = (float*)alloc((size_t)BN * DD * 4);

  auto cvt = [&](const float* s, bf16* d, int n) {
    f2bf_kernel<<<(n + 255) / 256, 256, 0, stream>>>(s, d, n);
  };
  cvt(so_w, sowB, SO_N * DD);
  cvt(aw_w, awwB, AW_N * DD);
  cvt(vp_w, vpwB, DD * DD);
  cvt(op_w, opwB, DD * DD);
  cvt(w1,   w1B,  DFFF * DD);
  cvt(w2,   w2B,  DD * DFFF);

  pos_embed_kernel<<<BN, 384, 0, stream>>>(x, qm, le, srcF, srcB, qposB);

  auto gemm = [&](const bf16* A, const bf16* Wt, const float* bias, int Mr,
                  int No, int K, float* oF, bf16* oB, int epi, const float* mk) {
    int grid = (Mr / 64) * ((No + 127) / 128);
    gemm_bf16_wmma<<<grid, 256, 0, stream>>>(A, Wt, bias, Mr, No, K, oF, oB, epi, mk);
  };

  // sampling offsets, attention logits, value projection (masked)
  gemm(qposB, sowB, so_b, BN, SO_N, DD, soF, nullptr, 0, nullptr);
  gemm(qposB, awwB, aw_b, BN, AW_N, DD, awF, nullptr, 0, nullptr);
  gemm(srcB,  vpwB, vp_b, BN, DD,   DD, valF, nullptr, 1, qm);

  softmax_p_kernel<<<(BN * MM + 255) / 256, 256, 0, stream>>>(awF);
  msda_gather_kernel<<<(BN * MM) / 8, 256, 0, stream>>>(valF, soF, awF, sampB);

  // output projection + LN1
  gemm(sampB, opwB, op_b, BN, DD, DD, attnF, nullptr, 0, nullptr);
  layernorm_kernel<<<BN, 384, 0, stream>>>(srcF, attnF, ln1_g, ln1_b, q1F, q1B, 0);

  // FFN + LN2 (final transposed store to [B,D,H,W])
  gemm(q1B, w1B, nullptr, BN, DFFF, DD,   nullptr, hB, 2, nullptr);
  gemm(hB,  w2B, nullptr, BN, DD,   DFFF, ffnF, nullptr, 0, nullptr);
  layernorm_kernel<<<BN, 384, 0, stream>>>(q1F, ffnF, ln2_g, ln2_b, out, nullptr, 1);
}
